// HGRNBitAttention_22935125361191
// MI455X (gfx1250) — 8-bit matrix operand order repaired and checked by reading, compiled, not run on hardware
//
#include <hip/hip_runtime.h>
#include <stdint.h>


typedef int v8i __attribute__((ext_vector_type(8)));

#define DDIM   1024
#define BB     4
#define TT     4096
#define TOKENS (BB * TT)
#define EPSF   1e-5f
#define NCHUNK 64
#define CLEN   (TT / NCHUNK)

__device__ __forceinline__ float sigf(float x) { return 1.0f / (1.0f + __expf(-x)); }

__device__ __forceinline__ void async_g2lds_b64(void* lds_generic, const void* gptr) {
  const uint32_t lds_addr = (uint32_t)(uintptr_t)lds_generic;
  asm volatile("global_load_async_to_lds_b64 %0, %1, off"
               :: "v"(lds_addr), "v"(gptr)
               : "memory");
}

__global__ void __launch_bounds__(256) k_wabs(const float* __restrict__ w,
                                              float* __restrict__ sum) {
  __shared__ float red[256];
  const int tid = threadIdx.x;
  const size_t n = (size_t)DDIM * DDIM;
  float s = 0.0f;
  for (size_t i = (size_t)blockIdx.x * blockDim.x + tid; i < n;
       i += (size_t)gridDim.x * blockDim.x)
    s += fabsf(w[i]);
  red[tid] = s;
  __syncthreads();
#pragma unroll
  for (int o = 128; o > 0; o >>= 1) {
    if (tid < o) red[tid] += red[tid + o];
    __syncthreads();
  }
  if (tid == 0) atomicAdd(sum, red[0]);
}

__global__ void __launch_bounds__(256) k_wquant(const float* __restrict__ w,
                                                const float* __restrict__ sum,
                                                signed char* __restrict__ q,
                                                float* __restrict__ wdeq) {
  const float mean = *sum * (1.0f / ((float)DDIM * (float)DDIM));
  const float m    = fmaxf(mean, 1e-5f);
  const float sc   = 1.0f / m;
  const size_t i = (size_t)blockIdx.x * blockDim.x + threadIdx.x;
  const float4 wv = ((const float4*)w)[i];
  int q0 = (int)fminf(fmaxf(rintf(wv.x * sc), -1.0f), 1.0f);
  int q1 = (int)fminf(fmaxf(rintf(wv.y * sc), -1.0f), 1.0f);
  int q2 = (int)fminf(fmaxf(rintf(wv.z * sc), -1.0f), 1.0f);
  int q3 = (int)fminf(fmaxf(rintf(wv.w * sc), -1.0f), 1.0f);
  uint32_t pk = (uint32_t)(uint8_t)(int8_t)q0 | ((uint32_t)(uint8_t)(int8_t)q1 << 8) |
                ((uint32_t)(uint8_t)(int8_t)q2 << 16) | ((uint32_t)(uint8_t)(int8_t)q3 << 24);
  ((uint32_t*)q)[i] = pk;
  if (i == 0) *wdeq = m;
}

__global__ void __launch_bounds__(256) k_actquant(const float* __restrict__ x,
                                                  const float* __restrict__ normw,
                                                  signed char* __restrict__ q,
                                                  float* __restrict__ deq) {
  __shared__ float red[256];
  const int t = blockIdx.x, tid = threadIdx.x;
  const float4 xv = ((const float4*)(x + (size_t)t * DDIM))[tid];
  red[tid] = xv.x * xv.x + xv.y * xv.y + xv.z * xv.z + xv.w * xv.w;
  __syncthreads();
#pragma unroll
  for (int o = 128; o > 0; o >>= 1) {
    if (tid < o) red[tid] += red[tid + o];
    __syncthreads();
  }
  const float rinv = rsqrtf(red[0] * (1.0f / (float)DDIM) + EPSF);
  __syncthreads();
  const float4 wv = ((const float4*)normw)[tid];
  const float n0 = xv.x * rinv * wv.x, n1 = xv.y * rinv * wv.y;
  const float n2 = xv.z * rinv * wv.z, n3 = xv.w * rinv * wv.w;
  red[tid] = fmaxf(fmaxf(fabsf(n0), fabsf(n1)), fmaxf(fabsf(n2), fabsf(n3)));
  __syncthreads();
#pragma unroll
  for (int o = 128; o > 0; o >>= 1) {
    if (tid < o) red[tid] = fmaxf(red[tid], red[tid + o]);
    __syncthreads();
  }
  const float mx = fmaxf(red[0], 1e-5f);
  const float sc = 127.0f / mx;
  int q0 = (int)fminf(fmaxf(rintf(n0 * sc), -128.0f), 127.0f);
  int q1 = (int)fminf(fmaxf(rintf(n1 * sc), -128.0f), 127.0f);
  int q2 = (int)fminf(fmaxf(rintf(n2 * sc), -128.0f), 127.0f);
  int q3 = (int)fminf(fmaxf(rintf(n3 * sc), -128.0f), 127.0f);
  uint32_t pk = (uint32_t)(uint8_t)(int8_t)q0 | ((uint32_t)(uint8_t)(int8_t)q1 << 8) |
                ((uint32_t)(uint8_t)(int8_t)q2 << 16) | ((uint32_t)(uint8_t)(int8_t)q3 << 24);
  ((uint32_t*)(q + (size_t)t * DDIM))[tid] = pk;
  if (tid == 0) deq[t] = mx * (1.0f / 127.0f);
}

__global__ void __launch_bounds__(256) k_gemm(const signed char* __restrict__ qa,
                                              const signed char* __restrict__ qw,
                                              const float* __restrict__ adeq,
                                              const float* __restrict__ wdeq,
                                              float* __restrict__ dst) {
  __shared__ __align__(16) unsigned char As[2][32 * 64];

  const int tid   = threadIdx.x;
  const int wave  = tid >> 5;
  const int lane  = tid & 31;
  const int waveM = wave >> 2;
  const int waveN = wave & 3;
  const int khalf = lane >> 4;
  const int l15   = lane & 15;
  const int m0    = blockIdx.x * 32;
  const int n0    = blockIdx.y * 256 + waveN * 64;

  v8i acc[4];
#pragma unroll
  for (int a = 0; a < 4; a++)
#pragma unroll
    for (int r = 0; r < 8; r++) acc[a][r] = 0;

  const int arow = tid >> 3;
  const int acol = (tid & 7) * 8;
  const signed char* aRow = qa + (size_t)(m0 + arow) * DDIM + acol;

  async_g2lds_b64(&As[0][arow * 64 + acol], aRow);

  const int NT = DDIM / 64;
  for (int t = 0; t < NT; ++t) {
    const int kk = t * 64;
    if (t + 1 < NT) {
      async_g2lds_b64(&As[(t + 1) & 1][arow * 64 + acol], aRow + (t + 1) * 64);
      asm volatile("s_wait_asynccnt 0x1" ::: "memory");
    } else {
      asm volatile("s_wait_asynccnt 0x0" ::: "memory");
    }
    __syncthreads();

    v8i afrag;
    const uint2* Arow = (const uint2*)(&As[t & 1][(waveM * 16 + l15) * 64]);
#pragma unroll
    for (int i = 0; i < 4; i++) {
      const uint2 v = Arow[(i >> 1) * 4 + khalf * 2 + (i & 1)];
      afrag[2 * i]     = (int)v.x;
      afrag[2 * i + 1] = (int)v.y;
    }

#pragma unroll
    for (int an = 0; an < 4; an++) {
      const signed char* wr = qw + (size_t)(n0 + an * 16 + l15) * DDIM + kk;
      if (an == 0 && kk + 64 < DDIM)
        __builtin_prefetch(wr + 64, 0, 1);
      const int4 lo = *(const int4*)(wr + khalf * 16);
      const int4 hi = *(const int4*)(wr + 32 + khalf * 16);
      v8i bfrag;
      bfrag[0] = lo.x; bfrag[1] = lo.y; bfrag[2] = lo.z; bfrag[3] = lo.w;
      bfrag[4] = hi.x; bfrag[5] = hi.y; bfrag[6] = hi.z; bfrag[7] = hi.w;
      acc[an] = __builtin_amdgcn_wmma_i32_16x16x64_iu8(
           true, afrag,  true, bfrag, acc[an],
           false,  false);
    }
    __syncthreads();
  }

  const float dw = *wdeq;
#pragma unroll
  for (int an = 0; an < 4; an++) {
#pragma unroll
    for (int r = 0; r < 8; r++) {
      const int m = m0 + waveM * 16 + khalf * 8 + r;
      const int n = n0 + an * 16 + l15;
      dst[(size_t)m * DDIM + n] = (float)acc[an][r] * adeq[m] * dw;
    }
  }
}

__global__ void __launch_bounds__(256) k_scan1(const float* __restrict__ fbuf,
                                               float* __restrict__ io,
                                               float* __restrict__ Pbuf,
                                               float* __restrict__ Hbuf) {
  const int bid   = blockIdx.x;
  const int dq    = bid & 3;
  const int chunk = (bid >> 2) & (NCHUNK - 1);
  const int b     = bid >> 8;
  const int d     = (dq << 8) + threadIdx.x;
  size_t idx = ((size_t)b * TT + (size_t)chunk * CLEN) * DDIM + d;
  float h = 0.0f, P = 1.0f;
  for (int t = 0; t < CLEN; t++, idx += DDIM) {
    const float a = sigf(fbuf[idx]);
    h = a * h + (1.0f - a) * io[idx];
    io[idx] = h;
    P *= a;
  }
  const size_t c = ((size_t)b * NCHUNK + chunk) * DDIM + d;
  Pbuf[c] = P;
  Hbuf[c] = h;
}

__global__ void __launch_bounds__(256) k_scan2(const float* __restrict__ Pbuf,
                                               float* __restrict__ Hbuf) {
  const int b = blockIdx.x >> 2;
  const int d = ((blockIdx.x & 3) << 8) + threadIdx.x;
  float h = 0.0f;
  size_t c = (size_t)b * NCHUNK * DDIM + d;
  for (int ch = 0; ch < NCHUNK; ch++, c += DDIM) {
    const float hl = Hbuf[c];
    const float P  = Pbuf[c];
    Hbuf[c] = h;
    h = P * h + hl;
  }
}

__global__ void __launch_bounds__(256) k_scan3(const float* __restrict__ fbuf,
                                               const float* __restrict__ Hbuf,
                                               float* __restrict__ io) {
  const int bid   = blockIdx.x;
  const int dq    = bid & 3;
  const int chunk = (bid >> 2) & (NCHUNK - 1);
  const int b     = bid >> 8;
  const int d     = (dq << 8) + threadIdx.x;
  const float Hinit = Hbuf[((size_t)b * NCHUNK + chunk) * DDIM + d];
  if (Hinit == 0.0f) return;
  size_t idx = ((size_t)b * TT + (size_t)chunk * CLEN) * DDIM + d;
  float P = 1.0f;
  for (int t = 0; t < CLEN; t++, idx += DDIM) {
    P *= sigf(fbuf[idx]);
    io[idx] += P * Hinit;
  }
}

__global__ void __launch_bounds__(256) k_gate(const float* __restrict__ o,
                                              const float* __restrict__ g,
                                              const float* __restrict__ gw,
                                              float* __restrict__ out) {
  __shared__ float red[256];
  const int t = blockIdx.x, tid = threadIdx.x;
  const float4 ov = ((const float4*)(o + (size_t)t * DDIM))[tid];
  red[tid] = ov.x * ov.x + ov.y * ov.y + ov.z * ov.z + ov.w * ov.w;
  __syncthreads();
#pragma unroll
  for (int off = 128; off > 0; off >>= 1) {
    if (tid < off) red[tid] += red[tid + off];
    __syncthreads();
  }
  const float rinv = rsqrtf(red[0] * (1.0f / (float)DDIM) + EPSF);
  const float4 gv = ((const float4*)(g + (size_t)t * DDIM))[tid];
  const float4 wv = ((const float4*)gw)[tid];
  float4 r;
  r.x = ov.x * rinv * wv.x * gv.x * sigf(gv.x);
  r.y = ov.y * rinv * wv.y * gv.y * sigf(gv.y);
  r.z = ov.z * rinv * wv.z * gv.z * sigf(gv.z);
  r.w = ov.w * rinv * wv.w * gv.w * sigf(gv.w);
  ((float4*)(out + (size_t)t * DDIM))[tid] = r;
}

extern "C" void kernel_launch(void* const* d_in, const int* in_sizes, int n_in,
                              void* d_out, int out_size, void* d_ws, size_t ws_size,
                              hipStream_t stream) {
  (void)in_sizes; (void)n_in; (void)out_size; (void)ws_size;

  const float* hs      = (const float*)d_in[0];
  const float* W[4]    = {(const float*)d_in[1], (const float*)d_in[2],
                          (const float*)d_in[3], (const float*)d_in[4]};
  const float* NW[4]   = {(const float*)d_in[5], (const float*)d_in[6],
                          (const float*)d_in[7], (const float*)d_in[8]};
  const float* g_norm_w = (const float*)d_in[9];

  char* ws = (char*)d_ws;
  float* wsum = (float*)ws;
  float* wdeq = wsum + 4;
  size_t off = 256;
  signed char* qact = (signed char*)(ws + off); off += (size_t)TOKENS * DDIM;
  float* adeq = (float*)(ws + off);             off += (size_t)TOKENS * 4;
  off = (off + 255) & ~(size_t)255;
  signed char* qw = (signed char*)(ws + off);   off += 4ull * DDIM * DDIM;
  off = (off + 255) & ~(size_t)255;
  const size_t actBytes = (size_t)TOKENS * DDIM * sizeof(float);
  float* ibuf = (float*)(ws + off); off += actBytes;
  float* fbuf = (float*)(ws + off); off += actBytes;
  float* gbuf = (float*)(ws + off); off += actBytes;
  float* Pbuf = (float*)(ws + off); off += (size_t)BB * NCHUNK * DDIM * 4;
  float* Hbuf = (float*)(ws + off); off += (size_t)BB * NCHUNK * DDIM * 4;

  const size_t WSTRIDE = (size_t)DDIM * DDIM;
  const dim3 gemmGrid(TOKENS / 32, DDIM / 256);
  const int scanBlocks = BB * NCHUNK * (DDIM / 256);

  hipMemsetAsync(wsum, 0, 4 * sizeof(float), stream);
  for (int s = 0; s < 4; s++)
    k_wabs<<<256, 256, 0, stream>>>(W[s], wsum + s);
  for (int s = 0; s < 4; s++)
    k_wquant<<<(DDIM * DDIM / 4) / 256, 256, 0, stream>>>(W[s], wsum + s,
                                                          qw + (size_t)s * WSTRIDE,
                                                          wdeq + s);

  k_actquant<<<TOKENS, 256, 0, stream>>>(hs, NW[0], qact, adeq);
  k_gemm<<<gemmGrid, 256, 0, stream>>>(qact, qw + 0 * WSTRIDE, adeq, wdeq + 0, ibuf);

  k_actquant<<<TOKENS, 256, 0, stream>>>(hs, NW[1], qact, adeq);
  k_gemm<<<gemmGrid, 256, 0, stream>>>(qact, qw + 1 * WSTRIDE, adeq, wdeq + 1, fbuf);

  k_actquant<<<TOKENS, 256, 0, stream>>>(hs, NW[2], qact, adeq);
  k_gemm<<<gemmGrid, 256, 0, stream>>>(qact, qw + 2 * WSTRIDE, adeq, wdeq + 2, gbuf);

  k_scan1<<<scanBlocks, 256, 0, stream>>>(fbuf, ibuf, Pbuf, Hbuf);
  k_scan2<<<BB * (DDIM / 256), 256, 0, stream>>>(Pbuf, Hbuf);
  k_scan3<<<scanBlocks, 256, 0, stream>>>(fbuf, Hbuf, ibuf);

  k_gate<<<TOKENS, 256, 0, stream>>>(ibuf, gbuf, g_norm_w, fbuf);
  k_actquant<<<TOKENS, 256, 0, stream>>>(fbuf, NW[3], qact, adeq);
  k_gemm<<<gemmGrid, 256, 0, stream>>>(qact, qw + 3 * WSTRIDE, adeq, wdeq + 3,
                                       (float*)d_out);
}
